// TriangleSelfAttention_34308198760743
// MI455X (gfx1250) — matrix operand order repaired and checked by reading, not compiled, not run on hardware
//
#include <hip/hip_runtime.h>
#include <hip/hip_bf16.h>
#include <math.h>

// ---------------------------------------------------------------------------
// TriangleSelfAttention on MI455X (gfx1250, wave32, WMMA 16x16x32 bf16)
//
// B=1, I=J=384, E=128, H=4, C=32, AHS=128.
// Memory-bound (~55 GFLOP vs ~600MB traffic @ 23.3TB/s):
//   - bf16 intermediates, fp32 WMMA accumulation
//   - flash-attention online softmax (scores never hit HBM)
//   - fragment-friendly layouts: q,k = [i][h][j][c], v = [i][h][c][j],
//     projection weights in ONE contiguous column-major bf16 buffer
//     (512 cols) so the GEMM inner loop has a single, branch-free
//     B-pointer (previous version generated saveexec-divergence there).
// ---------------------------------------------------------------------------

typedef __bf16 bf16_t;
typedef __attribute__((ext_vector_type(16))) __bf16 v16bf;
typedef __attribute__((ext_vector_type(8)))  float  v8f;

#define E_DIM   128
#define H_HEADS 4
#define C_DIM   32
#define AHS_DIM 128
#define SEQ     384
#define NROWS   (SEQ * SEQ)          // 147456 flattened (i,j) rows
#define NQKV    384                   // 3*H*C columns of Wqkv
#define NPROJ   512                   // 384 qkv + 128 gate columns
#define ATT_SCALE 0.17677669529663687f  // 1/sqrt(32)

union Frag {
    v16bf  v;
    __bf16 e[16];
    uint4  q[2];
};

__device__ __forceinline__ v8f zero8() {
    v8f z;
#pragma unroll
    for (int i = 0; i < 8; ++i) z[i] = 0.0f;
    return z;
}

__device__ __forceinline__ v8f wmma_bf16(v16bf a, v16bf b, v8f c) {
    // D = A(16x32 bf16) * B(32x16 bf16) + C(16x16 f32)
    return __builtin_amdgcn_wmma_f32_16x16x32_bf16(
        /*neg_a=*/false, a, /*neg_b=*/false, b,
        /*c_mod=*/(short)0, c, /*reuse_a=*/false, /*reuse_b=*/false);
}

// ---------------------------------------------------------------------------
// Kernel 1: weights -> bf16, column-major (B-fragment order).
//   wproj_t[col][k], col in [0,512): col<384 from Wqkv, col>=384 from Wg.
//   wo_t  [col][k] = Wo[k][col]
// ---------------------------------------------------------------------------
__global__ void wconv_kernel(const float* __restrict__ Wqkv,
                             const float* __restrict__ Wg,
                             const float* __restrict__ Wo,
                             bf16_t* __restrict__ wproj_t,
                             bf16_t* __restrict__ wo_t) {
    int t = blockIdx.x * blockDim.x + threadIdx.x;
    const int NP = NPROJ * E_DIM;      // 65536
    const int NO = E_DIM * E_DIM;      // 16384
    if (t < NP) {
        int col = t >> 7, k = t & 127;
        float v = (col < NQKV) ? Wqkv[k * NQKV + col]
                               : Wg[k * E_DIM + (col - NQKV)];
        wproj_t[t] = (bf16_t)v;
    } else if (t < NP + NO) {
        int u = t - NP; int col = u >> 7, k = u & 127;
        wo_t[u] = (bf16_t)Wo[k * E_DIM + col];
    }
}

// ---------------------------------------------------------------------------
// Kernel 2: LayerNorm over E=128, one wave (32 lanes x 4 elems) per row,
// output bf16 z_norm. Pure streaming kernel: prefetch ahead.
// ---------------------------------------------------------------------------
__global__ void ln_kernel(const float* __restrict__ z,
                          const float* __restrict__ lng,
                          const float* __restrict__ lnb,
                          bf16_t* __restrict__ zn) {
    int gtid = blockIdx.x * blockDim.x + threadIdx.x;
    int row  = gtid >> 5;
    int lane = gtid & 31;
    if (row >= NROWS) return;
    const float* zr = z + (size_t)row * E_DIM;
    if (row + 16 < NROWS)
        __builtin_prefetch(zr + 16 * E_DIM + lane * 4, 0, 0);  // global_prefetch_b8
    float4 x = *(const float4*)(zr + lane * 4);
    float s  = x.x + x.y + x.z + x.w;
    float s2 = x.x * x.x + x.y * x.y + x.z * x.z + x.w * x.w;
#pragma unroll
    for (int o = 16; o > 0; o >>= 1) {
        s  += __shfl_xor(s,  o, 32);
        s2 += __shfl_xor(s2, o, 32);
    }
    float mu  = s  * (1.0f / 128.0f);
    float var = s2 * (1.0f / 128.0f) - mu * mu;
    float rs  = rsqrtf(var + 1e-5f);
    float4 gg = *(const float4*)(lng + lane * 4);
    float4 bb = *(const float4*)(lnb + lane * 4);
    union { bf16_t h[4]; uint2 u; } o4;
    o4.h[0] = (bf16_t)((x.x - mu) * rs * gg.x + bb.x);
    o4.h[1] = (bf16_t)((x.y - mu) * rs * gg.y + bb.y);
    o4.h[2] = (bf16_t)((x.z - mu) * rs * gg.z + bb.z);
    o4.h[3] = (bf16_t)((x.w - mu) * rs * gg.w + bb.w);
    *(uint2*)(zn + (size_t)row * E_DIM + lane * 4) = o4.u;
}

// ---------------------------------------------------------------------------
// Kernel 3: fused QKV + gate projection.
// Block = one 16-row tile; 8 waves cover the 512 output columns, each wave a
// 16x64 tile (4 f32 accumulators), K=128 in 4 WMMA steps. Branch-free inner
// loop: single weight buffer, hoisted base pointers, immediate offsets.
// Scatter: q,k -> [i][h][j][c]; v -> [i][h][c][j]; gate -> sigmoid, bf16.
// ---------------------------------------------------------------------------
__global__ void qkvg_kernel(const bf16_t* __restrict__ zn,
                            const bf16_t* __restrict__ wproj_t,
                            const float*  __restrict__ bg,
                            bf16_t* __restrict__ qm,
                            bf16_t* __restrict__ km,
                            bf16_t* __restrict__ vm,
                            bf16_t* __restrict__ gm) {
    int lane = threadIdx.x & 31;
    int wave = threadIdx.x >> 5;          // 0..7
    int row0 = blockIdx.x * 16;
    int n_base = wave * 64;               // columns [n_base, n_base+64)
    int mrow = lane & 15;
    int half = lane >> 4;                 // 0/1: lane group

    v8f acc[4];
#pragma unroll
    for (int a = 0; a < 4; ++a) acc[a] = zero8();

    // Hoisted per-lane base pointers
    const bf16_t* arow = zn + (size_t)(row0 + mrow) * E_DIM + half * 8;
    const bf16_t* wb[4];
#pragma unroll
    for (int a = 0; a < 4; ++a)
        wb[a] = wproj_t + (size_t)(n_base + a * 16 + mrow) * E_DIM + half * 8;

#pragma unroll
    for (int kk = 0; kk < E_DIM; kk += 32) {
        Frag af;
        af.q[0] = *(const uint4*)(arow + kk);        // K 0..7  (g0) / 8..15  (g1)
        af.q[1] = *(const uint4*)(arow + kk + 16);   // K 16..23(g0) / 24..31 (g1)
#pragma unroll
        for (int a = 0; a < 4; ++a) {
            Frag bf_;
            bf_.q[0] = *(const uint4*)(wb[a] + kk);
            bf_.q[1] = *(const uint4*)(wb[a] + kk + 16);
            acc[a] = wmma_bf16(af.v, bf_.v, acc[a]);
        }
    }

#pragma unroll
    for (int a = 0; a < 4; ++a) {
        int col = n_base + a * 16 + mrow;
#pragma unroll
        for (int r = 0; r < 8; ++r) {
            int row = row0 + r + half * 8;
            int i = row / SEQ, j = row % SEQ;
            float val = acc[a][r];
            if (col < NQKV) {                 // wave-uniform in practice
                int s = col >> 7;             // 0=q 1=k 2=v
                int rem = col & 127;
                int h = rem >> 5, c = rem & 31;
                if (s == 0) {
                    qm[((size_t)(i * H_HEADS + h) * SEQ + j) * C_DIM + c] = (bf16_t)val;
                } else if (s == 1) {
                    km[((size_t)(i * H_HEADS + h) * SEQ + j) * C_DIM + c] = (bf16_t)val;
                } else {
                    vm[((size_t)(i * H_HEADS + h) * C_DIM + c) * SEQ + j] = (bf16_t)val;
                }
            } else {
                int gc = col - NQKV;
                float gv = val + bg[gc];
                gv = 1.0f / (1.0f + __expf(-gv));
                gm[(size_t)row * AHS_DIM + gc] = (bf16_t)gv;
            }
        }
    }
}

// ---------------------------------------------------------------------------
// Kernel 4: flash attention per (i, h). One wave per 16-q-row tile,
// 4 waves/block. Per 32-wide k block:
//   2x v_wmma (S = Q K^T, K-dim = C = 32)
//   online softmax (row = 16 lanes of one VGPR -> width-16 shfl reductions)
//   P transposed C-layout -> A-layout via per-wave LDS tile (1KB)
//   2x v_wmma (O += P V, K-dim = 32 k-positions)
// All loop-invariant base pointers hoisted; loop offsets are linear.
// ---------------------------------------------------------------------------
__global__ void attn_kernel(const bf16_t* __restrict__ qm,
                            const bf16_t* __restrict__ km,
                            const bf16_t* __restrict__ vm,
                            const unsigned char* __restrict__ zmask,
                            bf16_t* __restrict__ attn) {
    __shared__ bf16_t lds[4][16 * 32];
    int lane = threadIdx.x & 31;
    int wave = threadIdx.x >> 5;          // 0..3
    int blk  = blockIdx.x;                // 0 .. 384*4*6-1
    int chunk = blk % 6;
    int ih    = blk / 6;                  // i*H + h
    int i = ih / H_HEADS;
    int h = ih % H_HEADS;
    int q0 = (chunk * 4 + wave) * 16;
    int mrow = lane & 15;
    int half = lane >> 4;
    bf16_t* pt = lds[wave];

    // Q A-fragment (16 rows x K=32 channels), persistent for the whole loop
    Frag qa;
    const bf16_t* qb = qm + ((size_t)ih * SEQ + q0 + mrow) * C_DIM + half * 8;
    qa.q[0] = *(const uint4*)(qb);
    qa.q[1] = *(const uint4*)(qb + 16);

    float m_run[8], l_run[8];
#pragma unroll
    for (int r = 0; r < 8; ++r) { m_run[r] = -3.0e38f; l_run[r] = 0.0f; }
    v8f O0 = zero8(), O1 = zero8();

    // Hoisted bases (per-lane, loop-invariant)
    const bf16_t* k0p = km + (size_t)ih * SEQ * C_DIM + (size_t)mrow * C_DIM + half * 8;
    const bf16_t* k1p = k0p + 16 * C_DIM;
    const bf16_t* v0p = vm + (size_t)ih * C_DIM * SEQ + (size_t)mrow * SEQ + half * 8;
    const bf16_t* v1p = v0p + 16 * SEQ;
    const unsigned char* mkp = zmask + (size_t)i * SEQ + mrow;
    const bf16_t* prd = pt + mrow * 32 + half * 8;   // P reload base (A-layout)
    bf16_t* pw0 = pt + (half * 8) * 32 + mrow;        // P write base row r=0

    for (int j0 = 0; j0 < SEQ; j0 += 32) {
        // K B-fragments: B[c(K=32), kpos(N=16)] -> lane holds c = 8h..8h+7 and 16+8h..16+8h+7
        Frag kb0, kb1;
        {
            const bf16_t* p0 = k0p + (size_t)j0 * C_DIM;
            const bf16_t* p1 = k1p + (size_t)j0 * C_DIM;
            kb0.q[0] = *(const uint4*)(p0);
            kb0.q[1] = *(const uint4*)(p0 + 16);
            kb1.q[0] = *(const uint4*)(p1);
            kb1.q[1] = *(const uint4*)(p1 + 16);
        }

        v8f S0 = wmma_bf16(qa.v, kb0.v, zero8());
        v8f S1 = wmma_bf16(qa.v, kb1.v, zero8());

        float bias0 = mkp[j0]      ? 0.0f : -1.0e9f;
        float bias1 = mkp[j0 + 16] ? 0.0f : -1.0e9f;

#pragma unroll
        for (int r = 0; r < 8; ++r) {
            float s0 = S0[r] * ATT_SCALE + bias0;
            float s1 = S1[r] * ATT_SCALE + bias1;
            // row max across the 16 lanes holding this row's 32 columns
            float mx = fmaxf(s0, s1);
            mx = fmaxf(mx, __shfl_xor(mx, 8, 16));
            mx = fmaxf(mx, __shfl_xor(mx, 4, 16));
            mx = fmaxf(mx, __shfl_xor(mx, 2, 16));
            mx = fmaxf(mx, __shfl_xor(mx, 1, 16));
            float mn   = fmaxf(m_run[r], mx);
            float corr = __expf(m_run[r] - mn);
            float p0 = __expf(s0 - mn);
            float p1 = __expf(s1 - mn);
            float rsm = p0 + p1;
            rsm += __shfl_xor(rsm, 8, 16);
            rsm += __shfl_xor(rsm, 4, 16);
            rsm += __shfl_xor(rsm, 2, 16);
            rsm += __shfl_xor(rsm, 1, 16);
            l_run[r] = l_run[r] * corr + rsm;
            m_run[r] = mn;
            O0[r] *= corr;
            O1[r] *= corr;
            pw0[r * 32]      = (bf16_t)p0;     // cols j0..j0+15
            pw0[r * 32 + 16] = (bf16_t)p1;     // cols j0+16..j0+31
        }

        // Reload P in A-layout (same wave: compiler orders via s_wait_dscnt)
        Frag pa;
        pa.q[0] = *(const uint4*)(prd);
        pa.q[1] = *(const uint4*)(prd + 16);

        // V B-fragments: v stored [i][h][c][j]; columns contiguous over kpos
        Frag vb0, vb1;
        {
            const bf16_t* p0 = v0p + j0;
            const bf16_t* p1 = v1p + j0;
            vb0.q[0] = *(const uint4*)(p0);
            vb0.q[1] = *(const uint4*)(p0 + 16);
            vb1.q[0] = *(const uint4*)(p1);
            vb1.q[1] = *(const uint4*)(p1 + 16);
        }

        O0 = wmma_bf16(pa.v, vb0.v, O0);
        O1 = wmma_bf16(pa.v, vb1.v, O1);
    }

    // Epilogue: normalize by row sums, store attn[(i,j), h*32+c] bf16
#pragma unroll
    for (int r = 0; r < 8; ++r) {
        int Mr = r + half * 8;
        float inv = 1.0f / l_run[r];
        size_t base = ((size_t)i * SEQ + q0 + Mr) * AHS_DIM + h * C_DIM;
        attn[base + mrow]      = (bf16_t)(O0[r] * inv);
        attn[base + 16 + mrow] = (bf16_t)(O1[r] * inv);
    }
}

// ---------------------------------------------------------------------------
// Kernel 5: out = (g * attn) @ Wo + bo, masked. Gate multiply fused into the
// A-fragment build as a packed bf16 vector op. Wave = 16x64 tile; 2 waves
// cover N=128; block = 4 row tiles x 2 col waves = 256 threads.
// ---------------------------------------------------------------------------
__global__ void out_kernel(const bf16_t* __restrict__ gm,
                           const bf16_t* __restrict__ attn,
                           const bf16_t* __restrict__ wo_t,
                           const float*  __restrict__ bo,
                           const unsigned char* __restrict__ zmask,
                           float* __restrict__ out) {
    int lane = threadIdx.x & 31;
    int wave = threadIdx.x >> 5;          // 0..7
    int row0 = (blockIdx.x * 4 + (wave >> 1)) * 16;
    int n0   = (wave & 1) * 64;
    int mrow = lane & 15;
    int half = lane >> 4;

    v8f acc[4];
#pragma unroll
    for (int a = 0; a < 4; ++a) acc[a] = zero8();

    const bf16_t* grow = gm   + (size_t)(row0 + mrow) * AHS_DIM + half * 8;
    const bf16_t* arow = attn + (size_t)(row0 + mrow) * AHS_DIM + half * 8;
    const bf16_t* wb[4];
#pragma unroll
    for (int a = 0; a < 4; ++a)
        wb[a] = wo_t + (size_t)(n0 + a * 16 + mrow) * AHS_DIM + half * 8;

#pragma unroll
    for (int kk = 0; kk < AHS_DIM; kk += 32) {
        Frag gg, at, pa;
        gg.q[0] = *(const uint4*)(grow + kk);
        gg.q[1] = *(const uint4*)(grow + kk + 16);
        at.q[0] = *(const uint4*)(arow + kk);
        at.q[1] = *(const uint4*)(arow + kk + 16);
        pa.v = gg.v * at.v;                       // packed bf16 multiply
#pragma unroll
        for (int a = 0; a < 4; ++a) {
            Frag bf_;
            bf_.q[0] = *(const uint4*)(wb[a] + kk);
            bf_.q[1] = *(const uint4*)(wb[a] + kk + 16);
            acc[a] = wmma_bf16(pa.v, bf_.v, acc[a]);
        }
    }

#pragma unroll
    for (int a = 0; a < 4; ++a) {
        int col = n0 + a * 16 + mrow;
        float bias = bo[col];
#pragma unroll
        for (int r = 0; r < 8; ++r) {
            int row = row0 + r + half * 8;
            float o = acc[a][r] + bias;
            out[(size_t)row * E_DIM + col] = zmask[row] ? o : 0.0f;
        }
    }
}

// ---------------------------------------------------------------------------
// Host-side launch. Inputs (setup_inputs order):
// 0:z 1:z_mask 2:ln_g 3:ln_b 4:Wqkv 5:Wg 6:bg 7:Wo 8:bo
// ---------------------------------------------------------------------------
extern "C" void kernel_launch(void* const* d_in, const int* in_sizes, int n_in,
                              void* d_out, int out_size, void* d_ws, size_t ws_size,
                              hipStream_t stream) {
    const float* z      = (const float*)d_in[0];
    const unsigned char* zmask = (const unsigned char*)d_in[1];  // jax bool = 1 byte
    const float* ln_g   = (const float*)d_in[2];
    const float* ln_b   = (const float*)d_in[3];
    const float* Wqkv   = (const float*)d_in[4];
    const float* Wg     = (const float*)d_in[5];
    const float* bg     = (const float*)d_in[6];
    const float* Wo     = (const float*)d_in[7];
    const float* bo     = (const float*)d_in[8];
    float* out          = (float*)d_out;

    // Workspace layout (bf16, ~227 MB total)
    char* ws = (char*)d_ws;
    size_t off = 0;
    const size_t SZ_ROW128 = (size_t)NROWS * 128 * sizeof(bf16_t);   // 37,748,736
    bf16_t* zn   = (bf16_t*)(ws + off); off += SZ_ROW128;            // z_norm
    bf16_t* qm   = (bf16_t*)(ws + off); off += SZ_ROW128;            // [i][h][j][c]
    bf16_t* km   = (bf16_t*)(ws + off); off += SZ_ROW128;            // [i][h][j][c]
    bf16_t* vm   = (bf16_t*)(ws + off); off += SZ_ROW128;            // [i][h][c][j]
    bf16_t* gm   = (bf16_t*)(ws + off); off += SZ_ROW128;            // sigmoid gate
    bf16_t* attn = (bf16_t*)(ws + off); off += SZ_ROW128;            // attention out
    bf16_t* wproj_t = (bf16_t*)(ws + off); off += (size_t)NPROJ * E_DIM * sizeof(bf16_t);
    bf16_t* wo_t    = (bf16_t*)(ws + off); off += (size_t)E_DIM * E_DIM * sizeof(bf16_t);

    // 1. weights -> bf16 column-major (single contiguous projection buffer)
    {
        int n = NPROJ * E_DIM + E_DIM * E_DIM;   // 81920
        wconv_kernel<<<(n + 255) / 256, 256, 0, stream>>>(
            Wqkv, Wg, Wo, wproj_t, wo_t);
    }
    // 2. LayerNorm: 147456 rows, 8 waves/block
    ln_kernel<<<NROWS / 8, 256, 0, stream>>>(z, ln_g, ln_b, zn);
    // 3. QKV + gate projection: one block per 16-row tile
    qkvg_kernel<<<NROWS / 16, 256, 0, stream>>>(
        zn, wproj_t, bg, qm, km, vm, gm);
    // 4. Flash attention: (i*H) * 6 chunks, 4 waves/block (one q-tile each)
    attn_kernel<<<SEQ * H_HEADS * 6, 128, 0, stream>>>(
        qm, km, vm, zmask, attn);
    // 5. Gated output projection + mask
    out_kernel<<<(NROWS / 16) / 4, 256, 0, stream>>>(
        gm, attn, wo_t, bo, zmask, out);
}
